// EF_42511586295882
// MI455X (gfx1250) — compile-verified, matrix operand order checked by reading, not run on hardware
//
#include <hip/hip_runtime.h>

// ---------------------------------------------------------------------------
// CDNA5 / gfx1250 MPNN potential.
// Only x[:,0,:] of the (N,16,32) tensor reaches the output -> l>0 channels,
// spherical harmonics, and 16/17 of the scatter traffic are dead-code
// eliminated. All GEMM-shaped math runs on v_wmma_f32_16x16x32_f16.
// B-fragments are pre-packed to f16 in WMMA lane layout once; edge kernels
// stage gathered rows through LDS with coalesced float4 loads, use a
// branch-free full-tile fast path, and 32-bit unsigned offsets so atomics
// use the SADDR+voffset addressing mode (no per-element 64-bit adds).
// ---------------------------------------------------------------------------

typedef __attribute__((ext_vector_type(16))) _Float16 v16h;
typedef __attribute__((ext_vector_type(8)))  _Float16 v8h;
typedef __attribute__((ext_vector_type(8)))  float    v8f;

#define CUTOFF_R 6.0f
#define C00 0.282095f   // Y_00 constant: the only surviving sh component

__device__ __forceinline__ v8f wmma16(v16h a, v16h b, v8f c){
  // 8 args: (neg_a, A, neg_b, B, c_mod, C, reuse_a, reuse_b)
  return __builtin_amdgcn_wmma_f32_16x16x32_f16(false, a, false, b,
                                                (short)0, c, false, false);
}

// ---- WMMA fragment helpers (wave32, ISA 7.12.2 layouts) -------------------
// A (16 M x 32 K f16): lane holds row (lane&15); lanes<16 hold K{0..7,16..23},
// lanes>=16 hold K{8..15,24..31}.  128-bit vectorized f32 tile load:
__device__ __forceinline__ v16h load_a_f32(const float* t, int stride){
  int lane = threadIdx.x & 31, row = lane & 15, kh = (lane >> 4) * 8;
  const float* p = t + row * stride + kh;
  float4 a0 = *(const float4*)(p);
  float4 a1 = *(const float4*)(p + 4);
  float4 a2 = *(const float4*)(p + 16);
  float4 a3 = *(const float4*)(p + 20);
  v16h a;
  a[0]=(_Float16)a0.x;  a[1]=(_Float16)a0.y;  a[2]=(_Float16)a0.z;  a[3]=(_Float16)a0.w;
  a[4]=(_Float16)a1.x;  a[5]=(_Float16)a1.y;  a[6]=(_Float16)a1.z;  a[7]=(_Float16)a1.w;
  a[8]=(_Float16)a2.x;  a[9]=(_Float16)a2.y;  a[10]=(_Float16)a2.z; a[11]=(_Float16)a2.w;
  a[12]=(_Float16)a3.x; a[13]=(_Float16)a3.y; a[14]=(_Float16)a3.z; a[15]=(_Float16)a3.w;
  return a;
}
// A from a 16x16 f32 tile (K padded 16->32 with zeros)
__device__ __forceinline__ v16h load_a16_f32(const float* t){
  int lane = threadIdx.x & 31, row = lane & 15, kh = (lane >> 4) * 8;
  const float* p = t + row * 16 + kh;
  float4 a0 = *(const float4*)(p);
  float4 a1 = *(const float4*)(p + 4);
  v16h a;
  a[0]=(_Float16)a0.x; a[1]=(_Float16)a0.y; a[2]=(_Float16)a0.z; a[3]=(_Float16)a0.w;
  a[4]=(_Float16)a1.x; a[5]=(_Float16)a1.y; a[6]=(_Float16)a1.z; a[7]=(_Float16)a1.w;
#pragma unroll
  for (int j = 0; j < 8; ++j) a[j + 8] = (_Float16)0.0f;
  return a;
}
// B (32 K x 16 N f16): lane -> column, K mapping mirrors A.  Used only by the
// one-time pre-pack kernel; hot kernels load packed fragments.
__device__ __forceinline__ v16h build_b_w(const float* w, int nb, int keff){
  int lane = threadIdx.x & 31, col = (lane & 15) + nb, kh = (lane >> 4) * 8;
  v16h b;
#pragma unroll
  for (int j = 0; j < 8; ++j){
    int k0 = kh + j, k1 = 16 + kh + j;
    b[j]     = (_Float16)(k0 < keff ? w[k0 * 32 + col] : 0.0f);
    b[j + 8] = (_Float16)(k1 < keff ? w[k1 * 32 + col] : 0.0f);
  }
  return b;
}
__device__ __forceinline__ v16h build_b_comb(const float* w1, const float* w2,
                                             int nb){
  int lane = threadIdx.x & 31, col = (lane & 15) + nb, kh = (lane >> 4) * 8;
  v16h b;
#pragma unroll
  for (int j = 0; j < 8; ++j){
    int k = kh + j;                         // always < 16
    b[j]     = (_Float16)(C00 * w1[k * 32 + col] + w2[k * 32 + col]);
    b[j + 8] = (_Float16)0.0f;
  }
  return b;
}
// packed B-fragment load: 32 bytes per lane (two b128 loads)
__device__ __forceinline__ v16h load_bf(const _Float16* bf, int idx){
  int lane = threadIdx.x & 31;
  return *(const v16h*)(bf + (unsigned)(idx * 32 + lane) * 16u);
}
// C/D (16x16 f32): lane -> col, VGPR i -> row i + 8*(lane>>4)
__device__ __forceinline__ v8f load_c_f32(const float* t, int stride, int nb){
  int lane = threadIdx.x & 31, col = (lane & 15) + nb, mb = (lane >> 4) * 8;
  v8f c;
#pragma unroll
  for (int i = 0; i < 8; ++i) c[i] = t[(mb + i) * stride + col];
  return c;
}

// ---------------------------------------------------------------------------
// K0: one-time B-fragment pre-pack (one wave).
// frag ids: 0/1 = C00*Wr1_0+Wr2_0, 2/3 = W1_0, 4/5 = W2_0,
//           6/7 = Wr1_1, 8/9 = W1_1, 10/11 = W2_1   (even=cols0-15, odd=16-31)
// ---------------------------------------------------------------------------
__global__ __launch_bounds__(32) void k_pack_b(
    const float* __restrict__ Wr1_0, const float* __restrict__ Wr2_0,
    const float* __restrict__ W1_0,  const float* __restrict__ W2_0,
    const float* __restrict__ Wr1_1, const float* __restrict__ W1_1,
    const float* __restrict__ W2_1,  _Float16* __restrict__ bf)
{
  int lane = threadIdx.x;
  _Float16* p = bf + (size_t)lane * 16;
  *(v16h*)(p + 0  * 32 * 16) = build_b_comb(Wr1_0, Wr2_0, 0);
  *(v16h*)(p + 1  * 32 * 16) = build_b_comb(Wr1_0, Wr2_0, 16);
  *(v16h*)(p + 2  * 32 * 16) = build_b_w(W1_0, 0, 32);
  *(v16h*)(p + 3  * 32 * 16) = build_b_w(W1_0, 16, 32);
  *(v16h*)(p + 4  * 32 * 16) = build_b_w(W2_0, 0, 32);
  *(v16h*)(p + 5  * 32 * 16) = build_b_w(W2_0, 16, 32);
  *(v16h*)(p + 6  * 32 * 16) = build_b_w(Wr1_1, 0, 16);
  *(v16h*)(p + 7  * 32 * 16) = build_b_w(Wr1_1, 16, 16);
  *(v16h*)(p + 8  * 32 * 16) = build_b_w(W1_1, 0, 32);
  *(v16h*)(p + 9  * 32 * 16) = build_b_w(W1_1, 16, 32);
  *(v16h*)(p + 10 * 32 * 16) = build_b_w(W2_1, 0, 32);
  *(v16h*)(p + 11 * 32 * 16) = build_b_w(W2_1, 16, 32);
}

// ---------------------------------------------------------------------------
// K1: edge pass 1 — radial basis, ZBL pair energy, gc = rad @ (C00*Wr1_0+Wr2_0),
//     scatter gc * embed[z[src]] into x0pre[dst].  One wave = 16 edges.
// ---------------------------------------------------------------------------
__global__ __launch_bounds__(32) void k_edge1(
    const float* __restrict__ pos, const int* __restrict__ zno,
    const int* __restrict__ dst, const int* __restrict__ src,
    const float* __restrict__ embed, const _Float16* __restrict__ bf,
    float* __restrict__ x0pre, float* __restrict__ eacc,
    _Float16* __restrict__ radbuf, int E)
{
  __shared__ float s_rad[16 * 16];
  __shared__ float s_emb[16 * 32];
  __shared__ int   s_dst[16];
  __shared__ int   s_erow[16];
  int lane = threadIdx.x;
  int e0 = blockIdx.x * 16;
  bool full = (e0 + 16) <= E;

  if (lane < 16){
    int e = e0 + lane;
    if (e < E){
      int d = dst[e], s = src[e];
      float dx = pos[s * 3 + 0] - pos[d * 3 + 0];
      float dy = pos[s * 3 + 1] - pos[d * 3 + 1];
      float dz = pos[s * 3 + 2] - pos[d * 3 + 2];
      float r = sqrtf(dx * dx + dy * dy + dz * dz + 1e-10f);
      r = fmaxf(r, 1e-4f);
      float u = fminf(r * (1.0f / CUTOFF_R), 1.0f - 1e-6f);
      float cut = (r < CUTOFF_R) ? __expf(-u * u / (1.0f - u * u)) : 0.0f;
      float t = fminf(fmaxf(2.0f * __expf(-r) - 1.0f, -1.0f), 1.0f);
      float th = acosf(t);
      _Float16 hr[16];
#pragma unroll
      for (int k = 0; k < 16; ++k){
        float rv = __cosf((float)k * th) * cut;
        s_rad[lane * 16 + k] = rv;
        hr[k] = (_Float16)rv;
      }
      v8h lo, hi;
#pragma unroll
      for (int j = 0; j < 8; ++j){ lo[j] = hr[j]; hi[j] = hr[j + 8]; }
      *(v8h*)(radbuf + (size_t)e * 16)     = lo;   // 16B aligned f16 spill
      *(v8h*)(radbuf + (size_t)e * 16 + 8) = hi;
      // ZBL pair energy -> per-dst-atom accumulator
      float zd = (float)zno[d], zs = (float)zno[s];
      float a = (0.8854f * 0.529177f) /
                (powf(zd, 0.23f) + powf(zs, 0.23f) + 1e-10f);
      float ra = r / a;
      float phi = 0.18175f * __expf(-3.1998f  * ra)
                + 0.50986f * __expf(-0.94229f * ra)
                + 0.28022f * __expf(-0.4029f  * ra)
                + 0.02817f * __expf(-0.20162f * ra);
      atomicAdd(&eacc[d], 0.5f * 14.399645f * zd * zs / r * phi * cut);
      s_dst[lane]  = d;
      s_erow[lane] = zno[s] * 32;
    } else {
#pragma unroll
      for (int k = 0; k < 16; ++k) s_rad[lane * 16 + k] = 0.0f;
      s_dst[lane]  = -1;
      s_erow[lane] = 0;
    }
  }
  __syncthreads();

  // stage gathered embedding rows into LDS (coalesced float4 loads)
#pragma unroll
  for (int t = 0; t < 4; ++t){
    int i = lane + t * 32;            // 0..127 float4 slots
    int row = i >> 3, off = (i & 7) * 4;
    *(float4*)(s_emb + row * 32 + off) =
        *(const float4*)(embed + (unsigned)(s_erow[row] + off));
  }
  __syncthreads();

  v16h A = load_a16_f32(s_rad);
  v8f zc = {};
  v8f g0 = wmma16(A, load_bf(bf, 0), zc);
  v8f g1 = wmma16(A, load_bf(bf, 1), zc);

  int col = lane & 15, mb = (lane >> 4) * 8;
  float p0[8], p1[8]; int dd[8]; unsigned off[8];
#pragma unroll
  for (int i = 0; i < 8; ++i){
    int m = mb + i;
    dd[i]  = s_dst[m];
    off[i] = (unsigned)(dd[i] * 32 + col);     // 32-bit element offset
    p0[i]  = g0[i] * s_emb[m * 32 + col];
    p1[i]  = g1[i] * s_emb[m * 32 + col + 16];
  }
  if (full){
#pragma unroll
    for (int i = 0; i < 8; ++i){
      atomicAdd(x0pre + off[i], p0[i]);
      atomicAdd(x0pre + off[i] + 16u, p1[i]);
    }
  } else {
#pragma unroll
    for (int i = 0; i < 8; ++i){
      if (dd[i] >= 0){
        atomicAdd(x0pre + off[i], p0[i]);
        atomicAdd(x0pre + off[i] + 16u, p1[i]);
      }
    }
  }
}

// ---------------------------------------------------------------------------
// K2: node MLP 0 — x0 = x0pre + (h*silu(h)) @ W2_0, h = x0pre @ W1_0.
// ---------------------------------------------------------------------------
__global__ __launch_bounds__(32) void k_node0(
    const float* __restrict__ x0pre, const _Float16* __restrict__ bf,
    float* __restrict__ x0out, int N)
{
  __shared__ float s_t[16 * 32];
  int lane = threadIdx.x;
  int n0 = blockIdx.x * 16;
  const float* tile = x0pre + (unsigned)n0 * 32u;

  v16h A = load_a_f32(tile, 32);
  v8f zc = {};
  v8f h0 = wmma16(A, load_bf(bf, 2), zc);
  v8f h1 = wmma16(A, load_bf(bf, 3), zc);

  int col = lane & 15, mb = (lane >> 4) * 8;
#pragma unroll
  for (int i = 0; i < 8; ++i){                 // h*silu(h) = h^2 * sigmoid(h)
    float v0 = h0[i], v1 = h1[i];
    s_t[(mb + i) * 32 + col]      = v0 * v0 / (1.0f + __expf(-v0));
    s_t[(mb + i) * 32 + col + 16] = v1 * v1 / (1.0f + __expf(-v1));
  }
  __syncthreads();

  v16h A2 = load_a_f32(s_t, 32);
  v8f c0 = load_c_f32(tile, 32, 0);
  v8f c1 = load_c_f32(tile, 32, 16);
  v8f d0 = wmma16(A2, load_bf(bf, 4), c0);
  v8f d1 = wmma16(A2, load_bf(bf, 5), c1);

  unsigned ob = (unsigned)((n0 + mb) * 32 + col);
#pragma unroll
  for (int i = 0; i < 8; ++i){
    x0out[ob + (unsigned)i * 32u]       = d0[i];
    x0out[ob + (unsigned)i * 32u + 16u] = d1[i];
  }
}

// ---------------------------------------------------------------------------
// K3: edge pass 2 — g = rad @ Wr1_1, scatter g * x0[src] into y0[dst].
// ---------------------------------------------------------------------------
__global__ __launch_bounds__(32) void k_edge2(
    const _Float16* __restrict__ radbuf, const int* __restrict__ dst,
    const int* __restrict__ src, const float* __restrict__ x0,
    const _Float16* __restrict__ bf, float* __restrict__ y0, int E)
{
  __shared__ float s_x[16 * 32];
  __shared__ int s_dst[16], s_srow[16];
  int lane = threadIdx.x;
  int e0 = blockIdx.x * 16;
  bool full = (e0 + 16) <= E;

  if (lane < 16){
    int e = e0 + lane;
    s_dst[lane]  = (e < E) ? dst[e] : -1;
    s_srow[lane] = (e < E) ? src[e] * 32 : 0;
  }
  __syncthreads();

  // stage gathered x0 rows into LDS (coalesced float4 loads)
#pragma unroll
  for (int t = 0; t < 4; ++t){
    int i = lane + t * 32;
    int row = i >> 3, off = (i & 7) * 4;
    *(float4*)(s_x + row * 32 + off) =
        *(const float4*)(x0 + (unsigned)(s_srow[row] + off));
  }

  // A fragment from f16 radial spill (one 16B load per lane)
  int row = lane & 15, kh = (lane >> 4) * 8;
  int e = e0 + row;
  v16h A;
  if (full || e < E){
    v8h rv = *(const v8h*)(radbuf + (size_t)e * 16 + kh);
#pragma unroll
    for (int j = 0; j < 8; ++j){ A[j] = rv[j]; A[j + 8] = (_Float16)0.0f; }
  } else {
#pragma unroll
    for (int j = 0; j < 16; ++j) A[j] = (_Float16)0.0f;
  }
  __syncthreads();

  v8f zc = {};
  v8f g0 = wmma16(A, load_bf(bf, 6), zc);
  v8f g1 = wmma16(A, load_bf(bf, 7), zc);

  int col = lane & 15, mb = (lane >> 4) * 8;
  float p0[8], p1[8]; int dd[8]; unsigned off[8];
#pragma unroll
  for (int i = 0; i < 8; ++i){
    int m = mb + i;
    dd[i]  = s_dst[m];
    off[i] = (unsigned)(dd[i] * 32 + col);
    p0[i]  = g0[i] * s_x[m * 32 + col];
    p1[i]  = g1[i] * s_x[m * 32 + col + 16];
  }
  if (full){
#pragma unroll
    for (int i = 0; i < 8; ++i){
      atomicAdd(y0 + off[i], p0[i]);
      atomicAdd(y0 + off[i] + 16u, p1[i]);
    }
  } else {
#pragma unroll
    for (int i = 0; i < 8; ++i){
      if (dd[i] >= 0){
        atomicAdd(y0 + off[i], p0[i]);
        atomicAdd(y0 + off[i] + 16u, p1[i]);
      }
    }
  }
}

// ---------------------------------------------------------------------------
// K4: node MLP 1 + readout — t = y0 + silu(y0@W1_1)@W2_1;
//     e_atom = t.w_out + b_out[z] + eacc; masked segment-sum into molecules.
// ---------------------------------------------------------------------------
__global__ __launch_bounds__(32) void k_node1(
    const float* __restrict__ y0, const _Float16* __restrict__ bf,
    const float* __restrict__ w_out, const float* __restrict__ b_out,
    const int* __restrict__ zno, const float* __restrict__ eacc,
    const float* __restrict__ atom_mask, const int* __restrict__ bseg,
    const float* __restrict__ batch_mask, float* __restrict__ out, int N)
{
  __shared__ float s_t[16 * 32];
  __shared__ float s_d[16 * 32];
  int lane = threadIdx.x;
  int n0 = blockIdx.x * 16;
  const float* tile = y0 + (unsigned)n0 * 32u;

  v16h A = load_a_f32(tile, 32);
  v8f zc = {};
  v8f h0 = wmma16(A, load_bf(bf, 8), zc);
  v8f h1 = wmma16(A, load_bf(bf, 9), zc);

  int col = lane & 15, mb = (lane >> 4) * 8;
#pragma unroll
  for (int i = 0; i < 8; ++i){                 // silu(h)
    float v0 = h0[i], v1 = h1[i];
    s_t[(mb + i) * 32 + col]      = v0 / (1.0f + __expf(-v0));
    s_t[(mb + i) * 32 + col + 16] = v1 / (1.0f + __expf(-v1));
  }
  __syncthreads();

  v16h A2 = load_a_f32(s_t, 32);
  v8f c0 = load_c_f32(tile, 32, 0);
  v8f c1 = load_c_f32(tile, 32, 16);
  v8f d0 = wmma16(A2, load_bf(bf, 10), c0);
  v8f d1 = wmma16(A2, load_bf(bf, 11), c1);

#pragma unroll
  for (int i = 0; i < 8; ++i){
    s_d[(mb + i) * 32 + col]      = d0[i];
    s_d[(mb + i) * 32 + col + 16] = d1[i];
  }
  __syncthreads();

  if (lane < 16){
    int n = n0 + lane;
    if (n < N){
      float acc = 0.0f;
#pragma unroll
      for (int f = 0; f < 32; ++f) acc += s_d[lane * 32 + f] * w_out[f];
      float e = (acc + b_out[zno[n]] + eacc[n]) * atom_mask[n];
      int sg = bseg[n];
      atomicAdd(&out[sg], e * batch_mask[sg]);
    }
  }
}

// ---------------------------------------------------------------------------
extern "C" void kernel_launch(void* const* d_in, const int* in_sizes, int n_in,
                              void* d_out, int out_size, void* d_ws, size_t ws_size,
                              hipStream_t stream)
{
  (void)n_in; (void)ws_size;
  const int*   zno   = (const int*)  d_in[0];
  const float* pos   = (const float*)d_in[1];
  const int*   dst   = (const int*)  d_in[2];
  const int*   src   = (const int*)  d_in[3];
  const int*   bseg  = (const int*)  d_in[4];
  // d_in[5] = batch_size scalar (== out_size, unused on device)
  const float* bmask = (const float*)d_in[6];
  const float* amask = (const float*)d_in[7];
  const float* embed = (const float*)d_in[8];
  const float* Wr1_0 = (const float*)d_in[9];
  const float* Wr2_0 = (const float*)d_in[10];
  const float* W1_0  = (const float*)d_in[11];
  const float* W2_0  = (const float*)d_in[12];
  const float* Wr1_1 = (const float*)d_in[13];
  const float* W1_1  = (const float*)d_in[14];
  const float* W2_1  = (const float*)d_in[15];
  const float* w_out = (const float*)d_in[16];
  const float* b_out = (const float*)d_in[17];

  int N = in_sizes[0];
  int E = in_sizes[2];

  char* ws = (char*)d_ws;
  _Float16* bfrag = (_Float16*)ws;  ws += (size_t)12 * 32 * 16 * sizeof(_Float16);
  float* x0pre = (float*)ws;        ws += (size_t)N * 32 * sizeof(float);
  float* x0buf = (float*)ws;        ws += (size_t)N * 32 * sizeof(float);
  float* ybuf  = (float*)ws;        ws += (size_t)N * 32 * sizeof(float);
  float* eacc  = (float*)ws;        ws += (size_t)N * sizeof(float);
  _Float16* radbuf = (_Float16*)ws;   // E*16 f16

  hipMemsetAsync(x0pre, 0, (size_t)N * 32 * sizeof(float), stream);
  hipMemsetAsync(ybuf,  0, (size_t)N * 32 * sizeof(float), stream);
  hipMemsetAsync(eacc,  0, (size_t)N * sizeof(float), stream);
  hipMemsetAsync(d_out, 0, (size_t)out_size * sizeof(float), stream);

  int eb = (E + 15) / 16;
  int nb = (N + 15) / 16;
  k_pack_b<<<1, 32, 0, stream>>>(Wr1_0, Wr2_0, W1_0, W2_0, Wr1_1, W1_1, W2_1,
                                 bfrag);
  k_edge1<<<eb, 32, 0, stream>>>(pos, zno, dst, src, embed, bfrag,
                                 x0pre, eacc, radbuf, E);
  k_node0<<<nb, 32, 0, stream>>>(x0pre, bfrag, x0buf, N);
  k_edge2<<<eb, 32, 0, stream>>>(radbuf, dst, src, x0buf, bfrag, ybuf, E);
  k_node1<<<nb, 32, 0, stream>>>(ybuf, bfrag, w_out, b_out, zno, eacc,
                                 amask, bseg, bmask, (float*)d_out, N);
}
